// SAGE_5557687681533
// MI455X (gfx1250) — matrix operand order repaired and checked by reading, not compiled, not run on hardware
//
#include <hip/hip_runtime.h>
#include <hip/hip_fp16.h>

#define N_NODES  50000
#define N_EDGES  800000
#define FEATS    128
#define N_CLASSES 47
#define M_TILES  (N_NODES / 16)   // 3125 exactly

typedef __attribute__((ext_vector_type(16))) _Float16 v16h;
typedef __attribute__((ext_vector_type(8)))  float    v8f;

// ---------------- utility: zero f32 buffer (grid-stride) ----------------
__global__ void zero_f32(float* __restrict__ p, long n) {
    long i = (long)blockIdx.x * blockDim.x + threadIdx.x;
    long stride = (long)gridDim.x * blockDim.x;
    for (; i < n; i += stride) p[i] = 0.0f;
}

// ---------------- degree count + reciprocal ----------------
__global__ void degree_count(const int* __restrict__ dst, float* __restrict__ cnt) {
    int e = blockIdx.x * blockDim.x + threadIdx.x;
    if (e < N_EDGES) unsafeAtomicAdd(&cnt[dst[e]], 1.0f);
}

__global__ void inv_degree(const float* __restrict__ cnt, float* __restrict__ inv) {
    int n = blockIdx.x * blockDim.x + threadIdx.x;
    if (n < N_NODES) inv[n] = 1.0f / fmaxf(cnt[n], 1.0f);
}

// ---------------- scatter-add: one wave per edge, lane = 4 feats --------
__global__ void scatter_accum(const int* __restrict__ src, const int* __restrict__ dst,
                              const float* __restrict__ h, float* __restrict__ accum) {
    int t = blockIdx.x * blockDim.x + threadIdx.x;   // N_EDGES*32 threads exactly
    int e    = t >> 5;
    int lane = t & 31;
    if (e >= N_EDGES) return;
    const float4 v = *(const float4*)(h + (size_t)src[e] * FEATS + lane * 4);
    float* a = accum + (size_t)dst[e] * FEATS + lane * 4;
    unsafeAtomicAdd(a + 0, v.x);
    unsafeAtomicAdd(a + 1, v.y);
    unsafeAtomicAdd(a + 2, v.z);
    unsafeAtomicAdd(a + 3, v.w);
}

// ------- finalize mean + quantize self & neigh activations to f16 -------
__global__ void finalize_quant(const float* __restrict__ h, const float* __restrict__ accum,
                               const float* __restrict__ inv,
                               _Float16* __restrict__ hq_s, _Float16* __restrict__ hq_n) {
    long t = (long)blockIdx.x * blockDim.x + threadIdx.x;   // (N_NODES*FEATS)/4 threads
    long total4 = (long)N_NODES * FEATS / 4;
    if (t >= total4) return;
    long base = t * 4;
    int  node = (int)(base / FEATS);
    float iv = inv[node];
    float4 hs = *(const float4*)(h + base);
    float4 ha = *(const float4*)(accum + base);
    _Float16* os = hq_s + base;
    _Float16* on = hq_n + base;
    os[0] = (_Float16)hs.x; os[1] = (_Float16)hs.y; os[2] = (_Float16)hs.z; os[3] = (_Float16)hs.w;
    on[0] = (_Float16)(ha.x * iv); on[1] = (_Float16)(ha.y * iv);
    on[2] = (_Float16)(ha.z * iv); on[3] = (_Float16)(ha.w * iv);
}

// ------- pack W [dk x dn] f32 row-major into per-lane B register image --
// B 32x16 f16 packing used here (A matches): lane L = col (L&15); lanes 0-15 hold K=kb..kb+15,
// lanes 16-31 hold K=kb+16..kb+31; element e of the v16h = K offset e.
__global__ void pack_weight(const float* __restrict__ W, _Float16* __restrict__ out,
                            int dk, int dn, int ntiles, int nslices) {
    int t = blockIdx.x * blockDim.x + threadIdx.x;   // ntiles*nslices*32 threads
    int total = ntiles * nslices * 32;
    if (t >= total) return;
    int lane = t & 31;
    int s  = (t >> 5) % nslices;
    int nt = (t >> 5) / nslices;
    int n  = nt * 16 + (lane & 15);
    int kbase = s * 32 + (lane >> 4) * 16;
    _Float16 vals[16];
#pragma unroll
    for (int e = 0; e < 16; ++e) {
        int k = kbase + e;
        vals[e] = (n < dn && k < dk) ? (_Float16)W[(size_t)k * dn + n] : (_Float16)0.0f;
    }
    _Float16* o = out + (size_t)t * 16;
#pragma unroll
    for (int e = 0; e < 16; ++e) o[e] = vals[e];
}

// ---------------- fused dual-GEMM + bias (+ReLU) via WMMA ----------------
// out[m][n] = hq_s[m][:] @ Ws + hq_n[m][:] @ Wn + bias[n]
// one wave per 16x16 output tile; blockDim = 32*ntiles, gridDim = M_TILES
__global__ void sage_gemm(const _Float16* __restrict__ hqs, const _Float16* __restrict__ hqn,
                          const _Float16* __restrict__ Bs, const _Float16* __restrict__ Bn,
                          const float* __restrict__ bias, float* __restrict__ out,
                          int dn, int ntiles, int relu) {
    int wave = threadIdx.x >> 5;
    int lane = threadIdx.x & 31;
    int ntile = wave;
    if (ntile >= ntiles) return;                 // wave-uniform
    int m0  = blockIdx.x * 16;
    int row = m0 + (lane & 15);
    int h8  = (lane >> 4) * 8;                   // half-wave K/M offset

    const _Float16* as = hqs + (size_t)row * FEATS;
    const _Float16* an = hqn + (size_t)row * FEATS;

    union AV { v16h v; struct { uint4 lo, hi; } q; };
    v8f acc = {};

#pragma unroll
    for (int s = 0; s < 4; ++s) {
        int kb = s * 32;
        AV a; AV b;
        // ---- self ----
        a.q.lo = *(const uint4*)(as + kb + h8 * 2);      // K = kb+2*h8 .. +7
        a.q.hi = *(const uint4*)(as + kb + h8 * 2 + 8);  // K = kb+2*h8+8 .. +7
        b.v    = *(const v16h*)(Bs + ((size_t)(ntile * 4 + s) * 32 + lane) * 16);
        acc = __builtin_amdgcn_wmma_f32_16x16x32_f16(false, a.v, false, b.v,
                                                     (short)0, acc, false, false);
        // ---- neigh ----
        a.q.lo = *(const uint4*)(an + kb + h8 * 2);
        a.q.hi = *(const uint4*)(an + kb + h8 * 2 + 8);
        b.v    = *(const v16h*)(Bn + ((size_t)(ntile * 4 + s) * 32 + lane) * 16);
        acc = __builtin_amdgcn_wmma_f32_16x16x32_f16(false, a.v, false, b.v,
                                                     (short)0, acc, false, false);
    }

    int nc = ntile * 16 + (lane & 15);
    if (nc < dn) {
        float bv = bias[nc];
#pragma unroll
        for (int r = 0; r < 8; ++r) {            // C layout: VGPR r -> M = h8 + r
            float v = acc[r] + bv;
            if (relu) v = fmaxf(v, 0.0f);
            out[(size_t)(m0 + h8 + r) * dn + nc] = v;
        }
    }
}

// ---------------------------------------------------------------------------
extern "C" void kernel_launch(void* const* d_in, const int* in_sizes, int n_in,
                              void* d_out, int out_size, void* d_ws, size_t ws_size,
                              hipStream_t stream) {
    const float* x   = (const float*)d_in[0];
    const int*   src = (const int*)d_in[1];
    const int*   dst = (const int*)d_in[2];
    const float* Ws[3] = { (const float*)d_in[3], (const float*)d_in[6], (const float*)d_in[9]  };
    const float* Wn[3] = { (const float*)d_in[4], (const float*)d_in[7], (const float*)d_in[10] };
    const float* bb[3] = { (const float*)d_in[5], (const float*)d_in[8], (const float*)d_in[11] };

    // ---- workspace carve (256-B aligned) ----
    size_t off = 0;
    char* base = (char*)d_ws;
    auto carve = [&](size_t bytes) -> void* {
        void* p = base + off;
        off += (bytes + 255) & ~(size_t)255;
        return p;
    };
    float*    cnt   = (float*)   carve((size_t)N_NODES * 4);
    float*    inv   = (float*)   carve((size_t)N_NODES * 4);
    float*    accum = (float*)   carve((size_t)N_NODES * FEATS * 4);
    _Float16* hqs   = (_Float16*)carve((size_t)N_NODES * FEATS * 2);
    _Float16* hqn   = (_Float16*)carve((size_t)N_NODES * FEATS * 2);
    float*    hbuf  = (float*)   carve((size_t)N_NODES * FEATS * 4);
    _Float16* BsP   = (_Float16*)carve((size_t)8 * 4 * 32 * 16 * 2);   // 32 KB
    _Float16* BnP   = (_Float16*)carve((size_t)8 * 4 * 32 * 16 * 2);

    // ---- degrees (once) ----
    zero_f32<<<256, 256, 0, stream>>>(cnt, (long)N_NODES);
    degree_count<<<(N_EDGES + 255) / 256, 256, 0, stream>>>(dst, cnt);
    inv_degree<<<(N_NODES + 255) / 256, 256, 0, stream>>>(cnt, inv);

    const long feat_elems = (long)N_NODES * FEATS;
    const float* hin = x;
    for (int l = 0; l < 3; ++l) {
        int dn = (l == 2) ? N_CLASSES : FEATS;
        int ntiles = (dn + 15) / 16;

        zero_f32<<<4096, 256, 0, stream>>>(accum, feat_elems);
        scatter_accum<<<(N_EDGES * 32) / 256, 256, 0, stream>>>(src, dst, hin, accum);
        finalize_quant<<<(int)((feat_elems / 4 + 255) / 256), 256, 0, stream>>>(
            hin, accum, inv, hqs, hqn);

        int pthreads = ntiles * 4 * 32;
        pack_weight<<<(pthreads + 255) / 256, 256, 0, stream>>>(Ws[l], BsP, FEATS, dn, ntiles, 4);
        pack_weight<<<(pthreads + 255) / 256, 256, 0, stream>>>(Wn[l], BnP, FEATS, dn, ntiles, 4);

        float* outp = (l == 2) ? (float*)d_out : hbuf;
        sage_gemm<<<M_TILES, 32 * ntiles, 0, stream>>>(hqs, hqn, BsP, BnP, bb[l], outp,
                                                       dn, ntiles, (l < 2) ? 1 : 0);
        hin = hbuf;
    }
}
